// LTIModel_69896297775461
// MI455X (gfx1250) — matrix operand order repaired and checked by reading, not compiled, not run on hardware
//
#include <hip/hip_runtime.h>
#include <cstdint>

// ---------------------------------------------------------------------------
// Sizes (fixed by the reference)
// ---------------------------------------------------------------------------
#define B_   256
#define T_   2048
#define IN_  64
#define NX_  16
#define NU_  8
#define H_   64
#define DT_  0.01f
#define CCH  32   // chunks per batch
#define LCH  64   // steps per chunk  (CCH*LCH == T_)
#define GDS  32   // gd row stride (floats): 16 g | 8 d | 8 zero pad -> 128B rows

typedef __attribute__((ext_vector_type(16))) _Float16 v16h;
typedef __attribute__((ext_vector_type(8)))  _Float16 v8h;
typedef __attribute__((ext_vector_type(8)))  float    v8f;

// ---------------------------------------------------------------------------
// WMMA helper:  D = A(16x32 f16) x B(32x16 f16) + C(16x16 f32)
// ---------------------------------------------------------------------------
__device__ __forceinline__ v8f wmma16(v16h a, v16h b, v8f c) {
  return __builtin_amdgcn_wmma_f32_16x16x32_f16(
      /*neg_a=*/false, a, /*neg_b=*/false, b,
      /*c_mod=*/(short)0, c, /*reuse_a=*/false, /*reuse_b=*/false);
}

// A-fragment (16x32 f16) built from a global fp32 row.
// Layout: lane = M (mod 16); lanes 0-15 hold K = 0..15,
// lanes 16-31 hold K = 16..31 (matches b_frag).
__device__ __forceinline__ v16h a_frag_g(const float* __restrict__ row,
                                         int cbase, int half) {
  const float4* p0 = (const float4*)(row + cbase + half * 16);
  const float4* p1 = (const float4*)(row + cbase + 8 + half * 16);
  float4 f0 = p0[0], f1 = p0[1], f2 = p1[0], f3 = p1[1];
  v16h a;
  a[0]=(_Float16)f0.x; a[1]=(_Float16)f0.y; a[2]=(_Float16)f0.z; a[3]=(_Float16)f0.w;
  a[4]=(_Float16)f1.x; a[5]=(_Float16)f1.y; a[6]=(_Float16)f1.z; a[7]=(_Float16)f1.w;
  a[8]=(_Float16)f2.x; a[9]=(_Float16)f2.y; a[10]=(_Float16)f2.z; a[11]=(_Float16)f2.w;
  a[12]=(_Float16)f3.x; a[13]=(_Float16)f3.y; a[14]=(_Float16)f3.z; a[15]=(_Float16)f3.w;
  return a;
}

// A-fragment from a row-major f16 LDS row (for layers 2/3).
__device__ __forceinline__ v16h a_frag_lds(const _Float16* row, int cbase, int half) {
  v8h lo = *(const v8h*)(row + cbase + half * 16);
  v8h hi = *(const v8h*)(row + cbase + 8 + half * 16);
  v16h a;
#pragma unroll
  for (int i = 0; i < 8; ++i) { a[i] = lo[i]; a[8 + i] = hi[i]; }
  return a;
}

// B-fragment (32x16 f16): lane = N (mod 16); lanes 0-15 hold K 0-15 of the
// K-tile, lanes 16-31 hold K 16-31; contiguous when weights are stored [N][K].
__device__ __forceinline__ v16h b_frag(const _Float16* W, int N, int kbase) {
  return *(const v16h*)(W + N * 64 + kbase);
}

// ---------------------------------------------------------------------------
// Prep: f16 transposed weights, Mh = I + dt*A_T, P = Mh^64 (repeated squaring)
// ws f16 weight blob layout (halves): W1t[64][64] | W2t[64][64] | W3t[16][64] | Wpt[32][64]
// Wpt cols 0-15 = dt*By_T, cols 16-23 = Duy_T, 24-31 = 0.
// ---------------------------------------------------------------------------
__global__ __launch_bounds__(256) void k_prep(
    const float* __restrict__ A_T, const float* __restrict__ By_T,
    const float* __restrict__ Duy_T, const float* __restrict__ W1,
    const float* __restrict__ W2, const float* __restrict__ W3,
    _Float16* __restrict__ wsW, float* __restrict__ Mh, float* __restrict__ P) {
  int tid = threadIdx.x;
  _Float16* W1t = wsW;
  _Float16* W2t = wsW + 4096;
  _Float16* W3t = wsW + 8192;
  _Float16* Wpt = wsW + 9216;
  for (int i = tid; i < 4096; i += 256) {
    int n = i >> 6, k = i & 63;
    W1t[i] = (_Float16)W1[k * 64 + n];
    W2t[i] = (_Float16)W2[k * 64 + n];
  }
  for (int i = tid; i < 1024; i += 256) {
    int n = i >> 6, k = i & 63;
    W3t[i] = (n == 0) ? (_Float16)W3[k] : (_Float16)0.0f;
  }
  for (int i = tid; i < 2048; i += 256) {
    int n = i >> 6, k = i & 63;
    float v = 0.0f;
    if (n < 16)      v = DT_ * By_T[k * 16 + n];
    else if (n < 24) v = Duy_T[k * 8 + (n - 16)];
    Wpt[i] = (_Float16)v;
  }
  __shared__ float sA[256];
  {
    int m = tid >> 4, n = tid & 15;
    float mh = ((m == n) ? 1.0f : 0.0f) + DT_ * A_T[tid];
    Mh[tid] = mh;
    sA[tid] = mh;
  }
  __syncthreads();
  for (int s = 0; s < 6; ++s) {           // Mh^(2^6) = Mh^64
    int m = tid >> 4, n = tid & 15;
    float acc = 0.0f;
#pragma unroll
    for (int k = 0; k < 16; ++k) acc = fmaf(sA[m * 16 + k], sA[k * 16 + n], acc);
    __syncthreads();
    sA[tid] = acc;
    __syncthreads();
  }
  P[tid] = sA[tid];
}

// ---------------------------------------------------------------------------
// Fused WMMA kernel: per 16-row tile of obs:
//   gd[row][0:16] = dt*y@By_T, gd[row][16:24] = y@Duy_T, [24:32] = 0
//   value[row]    = tanh-MLP (64->64->64->1), f16 WMMA, f32 accum
// 8 waves/block, 1 tile/wave; weights staged in LDS once per block.
// ---------------------------------------------------------------------------
__global__ __launch_bounds__(256) void k_gemm(
    const float* __restrict__ obs, const float* __restrict__ b1,
    const float* __restrict__ b2, const float* __restrict__ b3,
    const _Float16* __restrict__ wsW, float* __restrict__ gd,
    float* __restrict__ value_out) {
  __shared__ alignas(32) _Float16 sWt[11264];       // 22 KB of f16 weights
  __shared__ alignas(32) _Float16 sH[8 * 16 * 64];  // per-wave activation tiles
  int tid = threadIdx.x;
  {
    const uint4* src = (const uint4*)wsW;
    uint4* dst = (uint4*)sWt;
    for (int i = tid; i < 1408; i += 256) dst[i] = src[i];
  }
  __syncthreads();
  const _Float16* sW1 = sWt;
  const _Float16* sW2 = sWt + 4096;
  const _Float16* sW3 = sWt + 8192;
  const _Float16* sWp = sWt + 9216;

  int wave = tid >> 5, lane = tid & 31;
  int M = lane & 15, half = lane >> 4;
  int rowbase = (blockIdx.x * 8 + wave) * 16;
  const float* orow = obs + (size_t)(rowbase + M) * IN_;

  v16h A0 = a_frag_g(orow, 0, half);
  v16h A1 = a_frag_g(orow, 32, half);

  // ---- projection: gd = obs @ Wp (32 cols, 24 real, 8 zero) ----
  v8f p0 = {}, p1 = {};
  p0 = wmma16(A0, b_frag(sWp, M, half * 16), p0);
  p0 = wmma16(A1, b_frag(sWp, M, 32 + half * 16), p0);
  p1 = wmma16(A0, b_frag(sWp, 16 + M, half * 16), p1);
  p1 = wmma16(A1, b_frag(sWp, 16 + M, 32 + half * 16), p1);
  // branch-free stores: cols 24-31 hold zeros (Wpt zero columns)
#pragma unroll
  for (int r = 0; r < 8; ++r) {
    size_t row = (size_t)(rowbase + r + 8 * half);
    gd[row * GDS + M] = p0[r];
    gd[row * GDS + 16 + M] = p1[r];
  }

  // ---- MLP layer 1: h = tanh(obs@W1 + b1) -> wave-private LDS tile ----
  _Float16* hbase = sH + wave * 1024;
#pragma unroll
  for (int nt = 0; nt < 4; ++nt) {
    v8f a = {};
    a = wmma16(A0, b_frag(sW1, nt * 16 + M, half * 16), a);
    a = wmma16(A1, b_frag(sW1, nt * 16 + M, 32 + half * 16), a);
    float bias = b1[nt * 16 + M];
#pragma unroll
    for (int r = 0; r < 8; ++r)
      hbase[(r + 8 * half) * 64 + nt * 16 + M] = (_Float16)tanhf(a[r] + bias);
  }

  // ---- layer 2 ----  (same-wave DS ops are in-order; no barrier needed)
  v16h H0 = a_frag_lds(hbase + M * 64, 0, half);
  v16h H1 = a_frag_lds(hbase + M * 64, 32, half);
#pragma unroll
  for (int nt = 0; nt < 4; ++nt) {
    v8f a = {};
    a = wmma16(H0, b_frag(sW2, nt * 16 + M, half * 16), a);
    a = wmma16(H1, b_frag(sW2, nt * 16 + M, 32 + half * 16), a);
    float bias = b2[nt * 16 + M];
#pragma unroll
    for (int r = 0; r < 8; ++r)
      hbase[(r + 8 * half) * 64 + nt * 16 + M] = (_Float16)tanhf(a[r] + bias);
  }

  // ---- layer 3: value = h2 @ W3 + b3 (W3 padded to 16 cols, col 0 real) ----
  v16h G0 = a_frag_lds(hbase + M * 64, 0, half);
  v16h G1 = a_frag_lds(hbase + M * 64, 32, half);
  v8f av = {};
  av = wmma16(G0, b_frag(sW3, M, half * 16), av);
  av = wmma16(G1, b_frag(sW3, M, 32 + half * 16), av);
  if (M == 0) {
    float bb = b3[0];
#pragma unroll
    for (int r = 0; r < 8; ++r)
      value_out[rowbase + r + 8 * half] = av[r] + bb;
  }
}

// ---------------------------------------------------------------------------
// Scan phase A: per (batch,chunk) lane, zero-init chunk end state
//   e = sum_j g_j @ Mh^(L-1-j)   (Mh wave-uniform -> scalar loads)
// ---------------------------------------------------------------------------
__global__ __launch_bounds__(256) void k_scanA(
    const float* __restrict__ gd, const float* __restrict__ Mh,
    float* __restrict__ e) {
  int gid = blockIdx.x * 256 + threadIdx.x;
  int b = gid >> 5, c = gid & 31;
  float Mr[256];
#pragma unroll
  for (int i = 0; i < 64; ++i) {
    float4 v = ((const float4*)Mh)[i];
    Mr[4 * i] = v.x; Mr[4 * i + 1] = v.y; Mr[4 * i + 2] = v.z; Mr[4 * i + 3] = v.w;
  }
  float x[16];
#pragma unroll
  for (int n = 0; n < 16; ++n) x[n] = 0.0f;
  const float* g = gd + ((size_t)b * T_ + (size_t)c * LCH) * GDS;
#pragma unroll 1
  for (int t = 0; t < LCH; ++t) {
    float nx[16];
#pragma unroll
    for (int i = 0; i < 4; ++i) {
      float4 v = ((const float4*)g)[i];
      nx[4 * i] = v.x; nx[4 * i + 1] = v.y; nx[4 * i + 2] = v.z; nx[4 * i + 3] = v.w;
    }
#pragma unroll
    for (int m = 0; m < 16; ++m) {
      float xm = x[m];
#pragma unroll
      for (int n = 0; n < 16; ++n) nx[n] = fmaf(xm, Mr[m * 16 + n], nx[n]);
    }
#pragma unroll
    for (int n = 0; n < 16; ++n) x[n] = nx[n];
    g += GDS;
  }
  float* ep = e + (size_t)gid * 16;
#pragma unroll
  for (int i = 0; i < 4; ++i)
    ((float4*)ep)[i] = make_float4(x[4 * i], x[4 * i + 1], x[4 * i + 2], x[4 * i + 3]);
}

// ---------------------------------------------------------------------------
// Scan phase Mid: per-batch sequential combine over chunks with P = Mh^L.
// Writes chunk start states and x_final.
// ---------------------------------------------------------------------------
__global__ __launch_bounds__(256) void k_scanMid(
    const float* __restrict__ x0, const float* __restrict__ P,
    const float* __restrict__ e, float* __restrict__ xs,
    float* __restrict__ xfinal) {
  int b = threadIdx.x;
  float Pr[256];
#pragma unroll
  for (int i = 0; i < 64; ++i) {
    float4 v = ((const float4*)P)[i];
    Pr[4 * i] = v.x; Pr[4 * i + 1] = v.y; Pr[4 * i + 2] = v.z; Pr[4 * i + 3] = v.w;
  }
  float x[16];
#pragma unroll
  for (int i = 0; i < 4; ++i) {
    float4 v = ((const float4*)(x0 + (size_t)b * 16))[i];
    x[4 * i] = v.x; x[4 * i + 1] = v.y; x[4 * i + 2] = v.z; x[4 * i + 3] = v.w;
  }
#pragma unroll 1
  for (int c = 0; c < CCH; ++c) {
    size_t idx = ((size_t)b * CCH + c) * 16;
#pragma unroll
    for (int i = 0; i < 4; ++i)
      ((float4*)(xs + idx))[i] =
          make_float4(x[4 * i], x[4 * i + 1], x[4 * i + 2], x[4 * i + 3]);
    float nx[16];
#pragma unroll
    for (int i = 0; i < 4; ++i) {
      float4 v = ((const float4*)(e + idx))[i];
      nx[4 * i] = v.x; nx[4 * i + 1] = v.y; nx[4 * i + 2] = v.z; nx[4 * i + 3] = v.w;
    }
#pragma unroll
    for (int m = 0; m < 16; ++m) {
      float xm = x[m];
#pragma unroll
      for (int n = 0; n < 16; ++n) nx[n] = fmaf(xm, Pr[m * 16 + n], nx[n]);
    }
#pragma unroll
    for (int n = 0; n < 16; ++n) x[n] = nx[n];
  }
#pragma unroll
  for (int i = 0; i < 4; ++i)
    ((float4*)(xfinal + (size_t)b * 16))[i] =
        make_float4(x[4 * i], x[4 * i + 1], x[4 * i + 2], x[4 * i + 3]);
}

// ---------------------------------------------------------------------------
// Scan phase B: replay each chunk from its start state, emit [u, log_stds].
// ---------------------------------------------------------------------------
__global__ __launch_bounds__(256) void k_scanB(
    const float* __restrict__ gd, const float* __restrict__ Mh,
    const float* __restrict__ CuT, const float* __restrict__ log_stds,
    const float* __restrict__ xs, float* __restrict__ outp) {
  int gid = blockIdx.x * 256 + threadIdx.x;
  int b = gid >> 5, c = gid & 31;
  float Mr[256];
#pragma unroll
  for (int i = 0; i < 64; ++i) {
    float4 v = ((const float4*)Mh)[i];
    Mr[4 * i] = v.x; Mr[4 * i + 1] = v.y; Mr[4 * i + 2] = v.z; Mr[4 * i + 3] = v.w;
  }
  float Cu[128];
#pragma unroll
  for (int i = 0; i < 32; ++i) {
    float4 v = ((const float4*)CuT)[i];
    Cu[4 * i] = v.x; Cu[4 * i + 1] = v.y; Cu[4 * i + 2] = v.z; Cu[4 * i + 3] = v.w;
  }
  float4 ls0 = ((const float4*)log_stds)[0];
  float4 ls1 = ((const float4*)log_stds)[1];
  float x[16];
#pragma unroll
  for (int i = 0; i < 4; ++i) {
    float4 v = ((const float4*)(xs + (size_t)gid * 16))[i];
    x[4 * i] = v.x; x[4 * i + 1] = v.y; x[4 * i + 2] = v.z; x[4 * i + 3] = v.w;
  }
  const float* g = gd + ((size_t)b * T_ + (size_t)c * LCH) * GDS;
  float* o = outp + ((size_t)b * T_ + (size_t)c * LCH) * 16;
#pragma unroll 1
  for (int t = 0; t < LCH; ++t) {
    float nx[16], dv[8];
#pragma unroll
    for (int i = 0; i < 4; ++i) {
      float4 v = ((const float4*)g)[i];
      nx[4 * i] = v.x; nx[4 * i + 1] = v.y; nx[4 * i + 2] = v.z; nx[4 * i + 3] = v.w;
    }
#pragma unroll
    for (int i = 0; i < 2; ++i) {
      float4 v = ((const float4*)g)[4 + i];
      dv[4 * i] = v.x; dv[4 * i + 1] = v.y; dv[4 * i + 2] = v.z; dv[4 * i + 3] = v.w;
    }
    float u[8];
#pragma unroll
    for (int j = 0; j < 8; ++j) u[j] = dv[j];
#pragma unroll
    for (int m = 0; m < 16; ++m) {
      float xm = x[m];
#pragma unroll
      for (int j = 0; j < 8; ++j) u[j] = fmaf(xm, Cu[m * 8 + j], u[j]);
    }
    ((float4*)o)[0] = make_float4(u[0], u[1], u[2], u[3]);
    ((float4*)o)[1] = make_float4(u[4], u[5], u[6], u[7]);
    ((float4*)o)[2] = ls0;
    ((float4*)o)[3] = ls1;
#pragma unroll
    for (int m = 0; m < 16; ++m) {
      float xm = x[m];
#pragma unroll
      for (int n = 0; n < 16; ++n) nx[n] = fmaf(xm, Mr[m * 16 + n], nx[n]);
    }
#pragma unroll
    for (int n = 0; n < 16; ++n) x[n] = nx[n];
    g += GDS;
    o += 16;
  }
}

// ---------------------------------------------------------------------------
extern "C" void kernel_launch(void* const* d_in, const int* in_sizes, int n_in,
                              void* d_out, int out_size, void* d_ws,
                              size_t ws_size, hipStream_t stream) {
  (void)in_sizes; (void)n_in; (void)out_size; (void)ws_size;
  const float* obs   = (const float*)d_in[0];
  const float* x0    = (const float*)d_in[1];
  const float* A_T   = (const float*)d_in[2];
  const float* By_T  = (const float*)d_in[3];
  const float* Cu_T  = (const float*)d_in[4];
  const float* Duy_T = (const float*)d_in[5];
  const float* lstd  = (const float*)d_in[6];
  const float* W1 = (const float*)d_in[7];
  const float* b1 = (const float*)d_in[8];
  const float* W2 = (const float*)d_in[9];
  const float* b2 = (const float*)d_in[10];
  const float* W3 = (const float*)d_in[11];
  const float* b3 = (const float*)d_in[12];

  float* outp  = (float*)d_out;                       // [B,T,16]
  float* value = outp + (size_t)B_ * T_ * 16;         // [B,T]
  float* xf    = value + (size_t)B_ * T_;             // [B,16]

  // workspace layout (~68 MB)
  float*    gd  = (float*)d_ws;                       // [B*T, 32] (24 used)
  _Float16* wsW = (_Float16*)(gd + (size_t)B_ * T_ * GDS);  // 11264 halves
  float*    Mh  = (float*)(wsW + 11264);              // [16,16]
  float*    P   = Mh + 256;                           // Mh^64
  float*    e   = P + 256;                            // [B*CCH,16]
  float*    xs  = e + (size_t)B_ * CCH * 16;          // [B*CCH,16]

  k_prep<<<1, 256, 0, stream>>>(A_T, By_T, Duy_T, W1, W2, W3, wsW, Mh, P);
  k_gemm<<<(B_ * T_) / 128, 256, 0, stream>>>(obs, b1, b2, b3, wsW, gd, value);
  k_scanA<<<(B_ * CCH) / 256, 256, 0, stream>>>(gd, Mh, e);
  k_scanMid<<<1, 256, 0, stream>>>(x0, P, e, xs, xf);
  k_scanB<<<(B_ * CCH) / 256, 256, 0, stream>>>(gd, Mh, Cu_T, lstd, xs, outp);
}
